// GnnBase_63969242906878
// MI455X (gfx1250) — matrix operand order repaired and checked by reading, not compiled, not run on hardware
//
#include <hip/hip_runtime.h>
#include <hip/hip_bf16.h>
#include <stdint.h>

// ---------------------------------------------------------------------------
// gso = knn-sparse correlation graph:
//   corr = Xn @ Xn^T (rows mean-centered, L2-normalized)
//   per row: keep the 81 smallest corr values, drop the minimum, scatter
//   ranks 1..80 (value at its column index) into a zeroed [N,N] matrix.
// N = 8192, D = 512, KNN = 80.
// ---------------------------------------------------------------------------

#define NN   8192
#define DD   512
#define KSEL 81          // keep 81 smallest, output entries 1..80
#define ASTRIDE 520      // padded LDS row stride (halves): +8 kills bank conflicts

typedef __attribute__((ext_vector_type(16))) _Float16 v16h;
typedef __attribute__((ext_vector_type(8)))  float    v8f;
typedef __attribute__((ext_vector_type(4)))  unsigned int v4u;

union AB { v16h h; v4u q[2]; };

__device__ __forceinline__ unsigned ballot32(bool p) {
#if __has_builtin(__builtin_amdgcn_ballot_w32)
    return (unsigned)__builtin_amdgcn_ballot_w32(p);
#else
    return (unsigned)__ballot(p);
#endif
}

// ---------------------------------------------------------------------------
// Kernel 0: zero the 256MB output (harness poisons it; sparse scatter later).
// ---------------------------------------------------------------------------
__global__ __launch_bounds__(256) void k_zero(float* __restrict__ out, long n4) {
    v4u* o4 = (v4u*)out;
    long i = (long)blockIdx.x * 256 + threadIdx.x;
    long stride = (long)gridDim.x * 256;
    v4u z = 0;  // splat
    for (; i < n4; i += stride) o4[i] = z;
}

// ---------------------------------------------------------------------------
// Kernel 1: row-normalize x (subtract mean, scale by 1/||xm||), emit f16.
// One wave per row (512 elems = 16 per lane), 8 waves per block.
// ---------------------------------------------------------------------------
__global__ __launch_bounds__(256) void k_normalize(const float* __restrict__ x,
                                                   _Float16* __restrict__ xn) {
    int wid  = threadIdx.x >> 5;
    int lane = threadIdx.x & 31;
    int row  = blockIdx.x * 8 + wid;
    const float* xr = x + (long)row * DD;

    float v[16];
    float s = 0.f;
#pragma unroll
    for (int j = 0; j < 16; ++j) { v[j] = xr[j * 32 + lane]; s += v[j]; }
#pragma unroll
    for (int o = 16; o; o >>= 1) s += __shfl_xor(s, o, 32);
    float mean = s * (1.0f / (float)DD);

    float sq = 0.f;
#pragma unroll
    for (int j = 0; j < 16; ++j) { v[j] -= mean; sq += v[j] * v[j]; }
#pragma unroll
    for (int o = 16; o; o >>= 1) sq += __shfl_xor(sq, o, 32);
    float inv = 1.0f / sqrtf(sq);

    _Float16* orow = xn + (long)row * DD;
#pragma unroll
    for (int j = 0; j < 16; ++j) orow[j * 32 + lane] = (_Float16)(v[j] * inv);
}

// ---------------------------------------------------------------------------
// Kernel 2: fused WMMA GEMM (16-row panel x all 8192 cols, K=512) + running
// 81-smallest selection per row + sparse scatter of ranks 1..80.
// Block: 512 threads (16 waves). Wave w computes the 16x16 tile at columns
// chunk*256 + w*16, then acts as the selector for panel row w.
// ---------------------------------------------------------------------------
__global__ __launch_bounds__(512) void k_gemm_select(const _Float16* __restrict__ xn,
                                                     float* __restrict__ out) {
    __shared__ __align__(16) _Float16 As[16 * ASTRIDE];   // padded A panel (f16)
    __shared__ float selbuf[16 * 256];                    // 16 rows x 256-col chunk
    __shared__ float topv[16][KSEL];
    __shared__ int   topi[16][KSEL];
    __shared__ int   cnt[16];

    const int tid     = threadIdx.x;
    const int w       = tid >> 5;        // wave id 0..15
    const int lane    = tid & 31;
    const int rowbase = blockIdx.x * 16;
    const int lhalf   = (lane < 16);     // low half-wave
    const int lm      = lane & 15;

    // --- load A panel (16 rows x 512 halves) into padded LDS ---
    {
        const v4u* src = (const v4u*)xn;   // 8 halves per v4u
#pragma unroll
        for (int t = 0; t < 2; ++t) {
            int i   = tid + t * 512;       // i in [0,1024): (row, j4) with 64 v4u/row
            int r   = i >> 6;
            int j4  = i & 63;
            ((v4u*)As)[r * (ASTRIDE / 8) + j4] = src[(long)(rowbase + r) * (DD / 8) + j4];
        }
    }
    for (int i = tid; i < 16 * KSEL; i += 512) {
        topv[i / KSEL][i % KSEL] = 3.0e38f;
        topi[i / KSEL][i % KSEL] = 0;
    }
    if (tid < 16) cnt[tid] = 0;
    __syncthreads();

    const v4u* xq = (const v4u*)xn;
    const v4u* aq = (const v4u*)As;
    // B-side: lane owns column n; B[k][n] = Xn[n][k] -> contiguous in k.
    const int  ncolbase = w * 16 + lm;
    // A-side per-lane v4u offsets (K-invariant parts)
    const int  a_lo = lm * (ASTRIDE / 8) + ((lhalf ? 0 : 8) >> 3);
    const int  a_hi = lm * (ASTRIDE / 8) + ((lhalf ? 16 : 24) >> 3);
    const int  bsel = (lhalf ? 0 : 8) >> 3;    // 0 or 1 v4u

    for (int chunk = 0; chunk < 32; ++chunk) {
        const int  ncol = chunk * 256 + ncolbase;
        const long brow = (long)ncol * (DD / 8);

        v8f c = {0.f, 0.f, 0.f, 0.f, 0.f, 0.f, 0.f, 0.f};
#pragma unroll 4
        for (int ks = 0; ks < 16; ++ks) {
            const int kb4 = ks * 4;        // K-step of 32 halves = 4 v4u
            AB a, b;
            a.q[0] = aq[a_lo + kb4];
            a.q[1] = aq[a_hi + kb4];
            b.q[0] = xq[brow + kb4 + bsel];
            b.q[1] = xq[brow + kb4 + bsel + 2];
            c = __builtin_amdgcn_wmma_f32_16x16x32_f16(
                    false, a.h, false, b.h, (short)0, c, false, false);
        }

        // dump tile: VGPR i -> row (i + 8*hi-half), col = w*16 + lane%16
#pragma unroll
        for (int i = 0; i < 8; ++i) {
            int m = i + (lhalf ? 0 : 8);
            selbuf[m * 256 + w * 16 + lm] = c[i];
        }
        __syncthreads();

        // --- selection: wave w scans row w's 256 fresh candidates ---
#pragma unroll 1
        for (int p = 0; p < 8; ++p) {
            int   pos  = p * 32 + lane;
            float val  = selbuf[w * 256 + pos];
            int   gcol = chunk * 256 + pos;
            int   c0   = cnt[w];
            float thr  = topv[w][KSEL - 1];
            bool  pass = (c0 < KSEL) || (val < thr);
            unsigned mask = ballot32(pass);
            while (mask) {
                int sb = __builtin_ctz(mask);
                mask &= mask - 1;
                float iv = __shfl(val, sb, 32);
                int   ic = __shfl(gcol, sb, 32);
                if (lane == 0) {           // serial sorted insert, re-checked
                    int cc = cnt[w];
                    if (cc == KSEL) {
                        if (iv < topv[w][KSEL - 1]) {
                            int j = KSEL - 1;
                            while (j > 0 && topv[w][j - 1] > iv) {
                                topv[w][j] = topv[w][j - 1];
                                topi[w][j] = topi[w][j - 1];
                                --j;
                            }
                            topv[w][j] = iv; topi[w][j] = ic;
                        }
                    } else {
                        int j = cc;
                        while (j > 0 && topv[w][j - 1] > iv) {
                            topv[w][j] = topv[w][j - 1];
                            topi[w][j] = topi[w][j - 1];
                            --j;
                        }
                        topv[w][j] = iv; topi[w][j] = ic;
                        cnt[w] = cc + 1;
                    }
                }
            }
        }
        __syncthreads();
    }

    // scatter ranks 1..80 (skip the single minimum, as reference does [1:81])
    {
        long rbase = (long)(rowbase + w) * NN;
        for (int e = 1 + lane; e < KSEL; e += 32)
            out[rbase + topi[w][e]] = topv[w][e];
    }
}

// ---------------------------------------------------------------------------
extern "C" void kernel_launch(void* const* d_in, const int* in_sizes, int n_in,
                              void* d_out, int out_size, void* d_ws, size_t ws_size,
                              hipStream_t stream) {
    (void)in_sizes; (void)n_in; (void)out_size; (void)ws_size;
    const float* x  = (const float*)d_in[0];
    float*       out = (float*)d_out;
    _Float16*    xn  = (_Float16*)d_ws;          // 8192*512 f16 = 8 MB scratch

    k_zero<<<8192, 256, 0, stream>>>(out, (long)NN * NN / 4);
    k_normalize<<<NN / 8, 256, 0, stream>>>(x, xn);
    k_gemm_select<<<NN / 16, 512, 0, stream>>>(xn, out);
}
